// SNAttentionBlock_19748259627398
// MI455X (gfx1250) — matrix operand order repaired and checked by reading, not compiled, not run on hardware
//
#include <hip/hip_runtime.h>
#include <hip/hip_bf16.h>

typedef __attribute__((ext_vector_type(16))) __bf16 v16bf;
typedef __attribute__((ext_vector_type(8)))  __bf16 v8bf;
typedef __attribute__((ext_vector_type(8)))  float  v8f;

constexpr int kB = 8, kC = 256, kN = 4096, kD = 32;
constexpr int kBR = 64;                     // queries per workgroup (attn)

static __device__ __forceinline__ v8f wmma_bf16(v16bf a, v16bf b, v8f c) {
  // D = A(16x32 bf16) * B(32x16 bf16) + C(16x16 f32)
  return __builtin_amdgcn_wmma_f32_16x16x32_bf16(false, a, false, b,
                                                 (short)0, c, false, false);
}

// Assemble a 16-element bf16 fragment from two 16-byte chunks.
static __device__ __forceinline__ v16bf frag16(const __bf16* lo, const __bf16* hi) {
  union { v16bf v; v8bf h[2]; } u;
  u.h[0] = *(const v8bf*)lo;
  u.h[1] = *(const v8bf*)hi;
  return u.v;
}

// ---------------------------------------------------------------------------
// Kernel 1: 1x1-conv projections.  Q[b,n,32], K[b,n,32] (row=position),
// V[b,c,n] — all bf16.  Grid (N/16, B), 160 threads = 5 waves, each wave owns
// 4 of the 20 output 16x16 tiles (2 Q d-tiles, 2 K d-tiles, 16 V c-tiles).
// ---------------------------------------------------------------------------
__global__ __launch_bounds__(160) void sn_proj_kernel(
    const float* __restrict__ x,
    const float* __restrict__ wq, const float* __restrict__ bq,
    const float* __restrict__ wk, const float* __restrict__ bk,
    const float* __restrict__ wv, const float* __restrict__ bv,
    __bf16* __restrict__ Qb, __bf16* __restrict__ Kb, __bf16* __restrict__ Vb)
{
  const int lane = threadIdx.x & 31;
  const int wave = threadIdx.x >> 5;          // 0..4
  const int l15  = lane & 15;
  const int hi   = lane >> 4;                 // 0 or 1
  const int b    = blockIdx.y;
  const int nb   = blockIdx.x * 16;
  const float* xb = x + (size_t)b * kC * kN;
  const int n    = nb + l15;                  // A-matrix row (spatial position)

  v8f acc[4];
  #pragma unroll
  for (int t = 0; t < 4; ++t) acc[t] = v8f{};

  #pragma unroll 1
  for (int kc = 0; kc < 8; ++kc) {            // C=256 in K=32 steps
    const int cb = kc * 32;
    // A fragment: X^T tile [16 n x 32 c].  Contiguous K layout:
    // lanes 0-15: elems 0-15 -> K0-15; lanes 16-31: K16-31.
    v16bf a;
    const int cA = cb + hi * 16;
    #pragma unroll
    for (int i = 0; i < 8; ++i)
      a[i] = (__bf16)xb[(size_t)(cA + i) * kN + n];
    #pragma unroll
    for (int i = 0; i < 8; ++i)
      a[8 + i] = (__bf16)xb[(size_t)(cA + 8 + i) * kN + n];

    #pragma unroll
    for (int t = 0; t < 4; ++t) {
      const int ot = wave * 4 + t;            // 0..19
      const float* wmat; int dbase;
      if (ot < 2)      { wmat = wq; dbase = ot * 16; }
      else if (ot < 4) { wmat = wk; dbase = (ot - 2) * 16; }
      else             { wmat = wv; dbase = (ot - 4) * 16; }
      // B fragment: W^T tile [32 c x 16 d]; per lane: column d = dbase+l15,
      // 16 contiguous c values (lanes 0-15: K0-15, lanes 16-31: K16-31).
      const float* wrow = wmat + (size_t)(dbase + l15) * kC + cb + hi * 16;
      v16bf bb;
      #pragma unroll
      for (int i = 0; i < 16; ++i) bb[i] = (__bf16)wrow[i];
      acc[t] = wmma_bf16(a, bb, acc[t]);
    }
  }

  // Store D tiles.  D layout: VGPR r -> row r (lanes 0-15) / r+8 (16-31),
  // col = lane&15.
  #pragma unroll
  for (int t = 0; t < 4; ++t) {
    const int ot = wave * 4 + t;
    if (ot < 4) {
      const int d     = (ot & 1) * 16 + l15;
      const float bia = (ot < 2) ? bq[d] : bk[d];
      __bf16* dst     = (ot < 2) ? Qb : Kb;
      #pragma unroll
      for (int r = 0; r < 8; ++r) {
        const int m = r + hi * 8;
        dst[((size_t)b * kN + nb + m) * kD + d] = (__bf16)(acc[t][r] + bia);
      }
    } else {
      const int co    = (ot - 4) * 16 + l15;
      const float bia = bv[co];
      #pragma unroll
      for (int r = 0; r < 8; ++r) {
        const int m = r + hi * 8;
        Vb[((size_t)b * kC + co) * kN + nb + m] = (__bf16)(acc[t][r] + bia);
      }
    }
  }
}

// ---------------------------------------------------------------------------
// Kernel 2: flash attention + epilogue.  Grid (N/64, B), 256 threads = 8
// waves, Br=64 queries per workgroup.  Waves 0-3 each own online softmax for
// one 16-query tile (4 concurrent softmax producers); all 8 waves consume the
// four P tiles from LDS and each accumulates a 64x32 f32 output block
// (4 q-tiles x 2 c-tiles of WMMA accumulators).
// ---------------------------------------------------------------------------
__global__ __launch_bounds__(256) void sn_attn_kernel(
    const float* __restrict__ x, const float* __restrict__ gammap,
    const __bf16* __restrict__ Qb, const __bf16* __restrict__ Kb,
    const __bf16* __restrict__ Vb, float* __restrict__ out)
{
  __shared__ __bf16 Plds[4][16 * 48];  // per q-tile: 16 rows x 32 keys, pad 48
  __shared__ float  sc_lds[kBR];       // per-row accumulator rescale
  __shared__ float  il_lds[kBR];       // per-row 1/l

  const int lane = threadIdx.x & 31;
  const int wave = threadIdx.x >> 5;   // 0..7
  const int l15  = lane & 15;
  const int hi   = lane >> 4;
  const int b    = blockIdx.y;
  const int qb0  = blockIdx.x * kBR;

  const __bf16* Qbat = Qb + (size_t)b * kN * kD;
  const __bf16* Kbat = Kb + (size_t)b * kN * kD;
  const __bf16* Vbat = Vb + (size_t)b * kC * kN;

  // Waves 0-3: softmax state for q-tile `wave` (rows r+8*hi per lane half;
  // identical across the 16 lanes of each half after xor-reductions).
  v16bf qfrag{};
  float m[8], l[8];
  if (wave < 4) {
    const __bf16* qp = Qbat + (size_t)(qb0 + wave * 16 + l15) * kD + hi * 16;
    qfrag = frag16(qp, qp + 8);
    #pragma unroll
    for (int r = 0; r < 8; ++r) { m[r] = -3.0e38f; l[r] = 0.f; }
  }

  const int c0 = wave * 32;
  v8f acc[4][2];
  #pragma unroll
  for (int qt = 0; qt < 4; ++qt) { acc[qt][0] = v8f{}; acc[qt][1] = v8f{}; }
  const __bf16* vrow0 = Vbat + (size_t)(c0 + l15) * kN;       // channel row
  const __bf16* vrow1 = vrow0 + (size_t)16 * kN;

  for (int jb = 0; jb < kN; jb += 32) {
    if (wave < 4) {
      // K^T B-fragments for key tiles [jb..jb+15], [jb+16..jb+31].
      const __bf16* kp0 = Kbat + (size_t)(jb + l15) * kD + hi * 16;
      const __bf16* kp1 = kp0 + 16 * kD;
      v16bf kf0 = frag16(kp0, kp0 + 8);
      v16bf kf1 = frag16(kp1, kp1 + 8);
      v8f e0 = wmma_bf16(qfrag, kf0, v8f{});
      v8f e1 = wmma_bf16(qfrag, kf1, v8f{});

      #pragma unroll
      for (int r = 0; r < 8; ++r) {
        // row max over 32 keys: reduce across the 16 lanes of each half
        float cmax = fmaxf(e0[r], e1[r]);
        #pragma unroll
        for (int off = 8; off >= 1; off >>= 1)
          cmax = fmaxf(cmax, __shfl_xor(cmax, off, 32));
        const float mn = fmaxf(m[r], cmax);
        const float s  = __expf(m[r] - mn);
        const float p0 = __expf(e0[r] - mn);
        const float p1 = __expf(e1[r] - mn);
        float rs = p0 + p1;
        #pragma unroll
        for (int off = 8; off >= 1; off >>= 1)
          rs += __shfl_xor(rs, off, 32);
        l[r] = l[r] * s + rs;
        m[r] = mn;
        const int rowm = r + hi * 8;
        Plds[wave][rowm * 48 + l15]      = (__bf16)p0;
        Plds[wave][rowm * 48 + 16 + l15] = (__bf16)p1;
        if (l15 == 0) sc_lds[wave * 16 + rowm] = s;
      }
    }
    __syncthreads();

    // V B-fragments: per lane, 16 contiguous keys on one channel row.
    const __bf16* vp0 = vrow0 + jb + hi * 16;
    const __bf16* vp1 = vrow1 + jb + hi * 16;
    v16bf vf0 = frag16(vp0, vp0 + 8);
    v16bf vf1 = frag16(vp1, vp1 + 8);
    if (jb + 32 < kN) {
      __builtin_prefetch(vrow0 + jb + 32, 0, 0);   // global_prefetch_b8
      __builtin_prefetch(vrow1 + jb + 32, 0, 0);
    }

    #pragma unroll
    for (int qt = 0; qt < 4; ++qt) {
      // rescale running accumulators by this q-tile's exp(m_old - m_new)
      #pragma unroll
      for (int r = 0; r < 8; ++r) {
        const float s = sc_lds[qt * 16 + r + hi * 8];
        acc[qt][0][r] *= s;
        acc[qt][1][r] *= s;
      }
      // P A-fragment from LDS (row = l15, K-chunks hi*16 / hi*16+8).
      const __bf16* pp = &Plds[qt][l15 * 48 + hi * 16];
      v16bf pf = frag16(pp, pp + 8);
      acc[qt][0] = wmma_bf16(pf, vf0, acc[qt][0]);
      acc[qt][1] = wmma_bf16(pf, vf1, acc[qt][1]);
    }
    __syncthreads();
  }

  if (wave < 4 && l15 == 0) {
    #pragma unroll
    for (int r = 0; r < 8; ++r) il_lds[wave * 16 + r + hi * 8] = 1.0f / l[r];
  }
  __syncthreads();

  // Epilogue: out = gamma * (acc / l) + x
  const float g = gammap[0];
  #pragma unroll
  for (int qt = 0; qt < 4; ++qt) {
    #pragma unroll
    for (int r = 0; r < 8; ++r) {
      const int rowm = r + hi * 8;
      const int row  = qb0 + qt * 16 + rowm;
      const float il = il_lds[qt * 16 + rowm];
      {
        const size_t idx = ((size_t)b * kC + c0 + l15) * kN + row;
        out[idx] = g * (acc[qt][0][r] * il) + x[idx];
      }
      {
        const size_t idx = ((size_t)b * kC + c0 + 16 + l15) * kN + row;
        out[idx] = g * (acc[qt][1][r] * il) + x[idx];
      }
    }
  }
}

// ---------------------------------------------------------------------------
extern "C" void kernel_launch(void* const* d_in, const int* in_sizes, int n_in,
                              void* d_out, int out_size, void* d_ws, size_t ws_size,
                              hipStream_t stream) {
  const float* x     = (const float*)d_in[0];
  const float* wq    = (const float*)d_in[1];
  const float* bq    = (const float*)d_in[2];
  const float* wk    = (const float*)d_in[3];
  const float* bk    = (const float*)d_in[4];
  const float* wv    = (const float*)d_in[5];
  const float* bv    = (const float*)d_in[6];
  const float* gamma = (const float*)d_in[7];
  float* out = (float*)d_out;

  char* ws = (char*)d_ws;
  __bf16* Qb = (__bf16*)ws;                                    // 2 MB
  __bf16* Kb = (__bf16*)(ws + (size_t)kB * kN * kD * 2);       // 2 MB
  __bf16* Vb = (__bf16*)(ws + (size_t)2 * kB * kN * kD * 2);   // 16 MB

  dim3 gproj(kN / 16, kB);
  sn_proj_kernel<<<gproj, 160, 0, stream>>>(x, wq, bq, wk, bk, wv, bv, Qb, Kb, Vb);
  dim3 gattn(kN / kBR, kB);
  sn_attn_kernel<<<gattn, 256, 0, stream>>>(x, gamma, Qb, Kb, Vb, out);
}
